// Dewarp_25675314495831
// MI455X (gfx1250) — matrix operand order repaired and checked by reading, not compiled, not run on hardware
//
#include <hip/hip_runtime.h>

typedef __attribute__((ext_vector_type(16))) _Float16 v16h;
typedef __attribute__((ext_vector_type(8)))  _Float16 v8h;
typedef __attribute__((ext_vector_type(8)))  float    v8f;

#define H_    1536
#define W_    1536
#define CH    3
#define RAD   6        // offsets dy,dx in [-6,6]  (D-1)
#define KS_   11
#define P_    5        // KS/2
#define TILE  16
#define TMPN  26       // TILE + 2*P_
#define AREG  38       // TMPN + 2*RAD

// ---------------------------------------------------------------------------
// One wave32 per 16x16 output tile. Per offset:
//   VALU: tmp = per-pixel 3-channel L2 norm on the 26x26 halo region
//   WMMA: 11x11 window sum = banded-ones matmuls (2 vertical + 1 horizontal)
//   VALU: branchless running argmin update
// All loops are exec-uniform; WMMA operand gathers are aligned b128 LDS loads
// from zero-padded f16 staging arrays.
// ---------------------------------------------------------------------------
__global__ __launch_bounds__(32)
void dewarp_main(const float* __restrict__ A, const float* __restrict__ Bm,
                 float* __restrict__ res)
{
    __shared__ float Alds[CH][AREG][AREG];        // A halo (zero outside image)
    __shared__ float Blds[CH][TMPN][TMPN];        // B halo (zero outside image)
    __shared__ float tmpC[TMPN][TMPN + 1];        // f32 tmp (exact center values)
    __shared__ __attribute__((aligned(16))) _Float16 tmpT[32][40]; // f16 tmp, [col][row]
    __shared__ __attribute__((aligned(16))) _Float16 VT[16][40];   // f16 vertical sums, [row][col]

    const int lane = threadIdx.x;                 // 0..31, wave32
    const int y0 = blockIdx.y * TILE;
    const int x0 = blockIdx.x * TILE;

    // ---- zero the f16 staging tile once: rows/cols >= 26 stay 0 forever ----
    {
        unsigned int* p = (unsigned int*)&tmpT[0][0];
        for (int i = lane; i < (32 * 40) / 2; i += 32) p[i] = 0u;
    }

    // ---- stage A halo into LDS (zero outside image == A_pad semantics) ----
    for (int idx = lane; idx < CH * AREG * AREG; idx += 32) {
        int c   = idx / (AREG * AREG);
        int rem = idx % (AREG * AREG);
        int r = rem / AREG, col = rem % AREG;
        int gy = y0 - (P_ + RAD) + r;
        int gx = x0 - (P_ + RAD) + col;
        float v = 0.f;
        if (gy >= 0 && gy < H_ && gx >= 0 && gx < W_)
            v = A[(size_t)c * H_ * W_ + (size_t)gy * W_ + gx];
        Alds[c][r][col] = v;
    }
    // ---- stage B halo into LDS ----
    for (int idx = lane; idx < CH * TMPN * TMPN; idx += 32) {
        int c   = idx / (TMPN * TMPN);
        int rem = idx % (TMPN * TMPN);
        int r = rem / TMPN, col = rem % TMPN;
        int gy = y0 - P_ + r;
        int gx = x0 - P_ + col;
        float v = 0.f;
        if (gy >= 0 && gy < H_ && gx >= 0 && gx < W_)
            v = Bm[(size_t)c * H_ * W_ + (size_t)gy * W_ + gx];
        Blds[c][r][col] = v;
    }
    __syncthreads();

    const int n  = lane & 15;
    const int hi = lane >> 4;

    // ---- offset-invariant banded-ones WMMA operands (registers) ----
    // a_band: A operand (16x32), Wband[m][K] = 1 iff m <= K <= m+10
    // b_band: B operand (32x16), W2[K][j]    = 1 iff j <= K <= j+10
    v16h a_band, b_band;
#pragma unroll
    for (int e = 0; e < 16; ++e) {
        int v = e >> 1, h = e & 1;
        int K = 16 * hi + 2 * v + h;
        a_band[e] = (K >= n && K <= n + (KS_ - 1)) ? (_Float16)1.0f : (_Float16)0.0f;
        int Kb = 16 * hi + e;
        b_band[e] = (Kb >= n && Kb <= n + (KS_ - 1)) ? (_Float16)1.0f : (_Float16)0.0f;
    }

    float resv[8], wrun[8];
#pragma unroll
    for (int v = 0; v < 8; ++v) { resv[v] = 0.f; wrun[v] = __builtin_inff(); }

    // reference order: dx outer, dy inner (tie-break: newer candidate wins)
    for (int o = 0; o < 169; ++o) {
        int dx = o / 13 - RAD;
        int dy = o % 13 - RAD;

        // ---- tmp region: branchless, exec-uniform (clamped tail) ----
#pragma unroll 1
        for (int t = 0; t < 22; ++t) {
            int idx = lane + 32 * t;
            idx = (idx < TMPN * TMPN) ? idx : (TMPN * TMPN - 1); // dup of last elem: benign
            int r = idx / TMPN, c = idx % TMPN;
            int gy = y0 - P_ + r, gx = x0 - P_ + c;
            int ar = r + dy + RAD;
            int ac = c + dx + RAD;
            float s = 0.f;
#pragma unroll
            for (int ch = 0; ch < CH; ++ch) {
                float d = Alds[ch][ar][ac] - Blds[ch][r][c];
                s = __builtin_fmaf(d, d, s);
            }
            bool valid = ((unsigned)gy < (unsigned)H_) & ((unsigned)gx < (unsigned)W_);
            float tval = valid ? __builtin_sqrtf(s) : 0.f;
            tmpC[r][c] = tval;               // exact f32 (center candidates)
            tmpT[c][r] = (_Float16)tval;     // f16 transposed (WMMA B operand)
        }
        __syncthreads();

        // ---- hoisted, unconditional center loads (hide latency under WMMA) ----
        float ctr[8];
#pragma unroll
        for (int v = 0; v < 8; ++v)
            ctr[v] = tmpC[v + 8 * hi + P_][n + P_];

        // ---- vertical pass: V = Wband x tmp   (two 16-column chunks) ----
        // B operand layout: lane n, K = 16*hi + e -> 16 consecutive halves.
        v8h b0lo = *(const v8h*)&tmpT[n][16 * hi];
        v8h b0hi = *(const v8h*)&tmpT[n][16 * hi + 8];
        v8h b1lo = *(const v8h*)&tmpT[16 + n][16 * hi];
        v8h b1hi = *(const v8h*)&tmpT[16 + n][16 * hi + 8];
        v16h tb0 = __builtin_shufflevector(b0lo, b0hi,
            0, 1, 2, 3, 4, 5, 6, 7, 8, 9, 10, 11, 12, 13, 14, 15);
        v16h tb1 = __builtin_shufflevector(b1lo, b1hi,
            0, 1, 2, 3, 4, 5, 6, 7, 8, 9, 10, 11, 12, 13, 14, 15);

        v8f cz = {};
        v8f Va = __builtin_amdgcn_wmma_f32_16x16x32_f16(
            false, a_band, false, tb0, (short)0, cz, false, false);
        v8f Vb = __builtin_amdgcn_wmma_f32_16x16x32_f16(
            false, a_band, false, tb1, (short)0, cz, false, false);

        // D layout: row = v + 8*hi, col = n (chunk1 -> cols 16..31).
        // Vb cols 26..31 are exactly 0 (their B columns were 0): no guards needed.
#pragma unroll
        for (int v = 0; v < 8; ++v) {
            VT[v + 8 * hi][n]      = (_Float16)Va[v];
            VT[v + 8 * hi][16 + n] = (_Float16)Vb[v];
        }
        __syncthreads();

        // ---- horizontal pass: S = V x W2 ----
        // A operand layout: lane row m = n; K = 16*hi + e:
        // 16 consecutive halves of VT row n.
        v8h alo = *(const v8h*)&VT[n][16 * hi];
        v8h ahi = *(const v8h*)&VT[n][16 * hi + 8];
        v16h av = __builtin_shufflevector(alo, ahi,
            0, 1, 2, 3, 4, 5, 6, 7, 8, 9, 10, 11, 12, 13, 14, 15);

        v8f S = __builtin_amdgcn_wmma_f32_16x16x32_f16(
            false, av, false, b_band, (short)0, cz, false, false);

        // ---- branchless running argmin (S = 121*w; positive scale is monotone) ----
#pragma unroll
        for (int v = 0; v < 8; ++v) {
            float w = S[v];
            bool take = (wrun[v] >= w);          // ties -> newer candidate, as reference
            resv[v] = take ? ctr[v] : resv[v];   // v_cndmask
            wrun[v] = take ? w      : wrun[v];   // v_cndmask
        }
        __syncthreads();   // tmpC/tmpT/VT reused next offset
    }

    // ---- write pre-minpool result ----
#pragma unroll
    for (int v = 0; v < 8; ++v) {
        int gy = y0 + v + 8 * hi;
        int gx = x0 + n;
        res[(size_t)gy * W_ + gx] = resv[v];
    }
}

// ---------------------------------------------------------------------------
// 3x3 min-pool with +inf padding (== -maxpool(-x))
// ---------------------------------------------------------------------------
__global__ __launch_bounds__(256)
void minpool3(const float* __restrict__ res, float* __restrict__ out)
{
    int idx = blockIdx.x * blockDim.x + threadIdx.x;
    if (idx >= H_ * W_) return;
    int y = idx / W_, x = idx % W_;
    float m = __builtin_inff();
#pragma unroll
    for (int dy = -1; dy <= 1; ++dy)
#pragma unroll
        for (int dx = -1; dx <= 1; ++dx) {
            int yy = y + dy, xx = x + dx;
            if (yy >= 0 && yy < H_ && xx >= 0 && xx < W_)
                m = fminf(m, res[(size_t)yy * W_ + xx]);
        }
    out[idx] = m;
}

// ---------------------------------------------------------------------------
extern "C" void kernel_launch(void* const* d_in, const int* in_sizes, int n_in,
                              void* d_out, int out_size, void* d_ws, size_t ws_size,
                              hipStream_t stream)
{
    const float* A  = (const float*)d_in[0];   // (1,3,1536,1536) fp32
    const float* Bm = (const float*)d_in[1];   // (1,3,1536,1536) fp32
    float* out = (float*)d_out;                // (1,1,1536,1536) fp32
    float* res = (float*)d_ws;                 // 1536*1536 fp32 scratch

    dim3 grid(W_ / TILE, H_ / TILE);           // 96 x 96 tiles
    dewarp_main<<<grid, 32, 0, stream>>>(A, Bm, res);

    int total = H_ * W_;
    minpool3<<<(total + 255) / 256, 256, 0, stream>>>(res, out);
}
